// ScaledDotProductAttention_53403623359074
// MI455X (gfx1250) — matrix operand order repaired and checked by reading, not compiled, not run on hardware
//
#include <hip/hip_runtime.h>

typedef __attribute__((ext_vector_type(16))) _Float16 v16h;
typedef __attribute__((ext_vector_type(8)))  _Float16 v8h;
typedef __attribute__((ext_vector_type(8)))  float    v8f;

#define B_DIM 4
#define H_DIM 16
#define S_LEN 2048
#define D_DIM 64
// exp2-domain score scale: (1/sqrt(64)) * log2(e)
#define SCALE2 (0.125f * 1.4426950408889634f)
#define NEGBIG -1.0e9f

// convert 8 consecutive f32 -> f16, store 16B to LDS
__device__ __forceinline__ void cvt_store8(_Float16* dst, const float* src) {
    float4 a = ((const float4*)src)[0];
    float4 b = ((const float4*)src)[1];
    v8h h;
    h[0] = (_Float16)a.x; h[1] = (_Float16)a.y;
    h[2] = (_Float16)a.z; h[3] = (_Float16)a.w;
    h[4] = (_Float16)b.x; h[5] = (_Float16)b.y;
    h[6] = (_Float16)b.z; h[7] = (_Float16)b.w;
    *(v8h*)dst = h;
}

__device__ __forceinline__ v8h cvt8(float4 a, float4 b) {
    v8h h;
    h[0] = (_Float16)a.x; h[1] = (_Float16)a.y;
    h[2] = (_Float16)a.z; h[3] = (_Float16)a.w;
    h[4] = (_Float16)b.x; h[5] = (_Float16)b.y;
    h[6] = (_Float16)b.z; h[7] = (_Float16)b.w;
    return h;
}

__device__ __forceinline__ v16h cat8(v8h a, v8h b) {
    v16h r;
    #pragma unroll
    for (int i = 0; i < 8; ++i) { r[i] = a[i]; r[i + 8] = b[i]; }
    return r;
}

__global__ __launch_bounds__(256)
void sdpa_causal_wmma_kernel(const float* __restrict__ Qg,
                             const float* __restrict__ Kg,
                             const float* __restrict__ Vg,
                             float* __restrict__ ctx_out,
                             float* __restrict__ prob_out) {
    // double-buffered staging: one barrier per k-tile
    __shared__ __align__(16) _Float16 sK [2][32][80];   // 10240 B
    __shared__ __align__(16) _Float16 sVt[2][64][40];   // 10240 B
    __shared__ __align__(16) _Float16 sP [8][16][48];   // 12288 B

    const int bh    = blockIdx.y;
    const int q0    = blockIdx.x * 128;
    const int tid   = threadIdx.x;
    const int wave  = tid >> 5;
    const int lane  = tid & 31;
    const int half  = lane >> 4;
    const int n16   = lane & 15;
    const int qrow0 = q0 + wave * 16;
    // wave-uniform causal bound as a scalar (SALU branch, EXEC untouched)
    const int qlast = __builtin_amdgcn_readfirstlane(qrow0 + 15);

    const float* Qp = Qg + (size_t)bh * S_LEN * D_DIM;
    const float* Kp = Kg + (size_t)bh * S_LEN * D_DIM;
    const float* Vp = Vg + (size_t)bh * S_LEN * D_DIM;
    float* ctxp  = ctx_out  + (size_t)bh * S_LEN * D_DIM;
    float* probp = prob_out + (size_t)bh * S_LEN * S_LEN;

    // ---- Q A-fragments (16x32 f16), held in VGPRs for the whole k sweep ----
    v16h aQ0, aQ1;
    {
        const float* qrow = Qp + (size_t)(qrow0 + n16) * D_DIM;
        const int c0 = 16 * half;
        const int c1 = 16 * half + 8;
        float4 a0 = *(const float4*)(qrow + c0);
        float4 a1 = *(const float4*)(qrow + c0 + 4);
        float4 a2 = *(const float4*)(qrow + c1);
        float4 a3 = *(const float4*)(qrow + c1 + 4);
        aQ0 = cat8(cvt8(a0, a1), cvt8(a2, a3));
        float4 b0 = *(const float4*)(qrow + 32 + c0);
        float4 b1 = *(const float4*)(qrow + 32 + c0 + 4);
        float4 b2 = *(const float4*)(qrow + 32 + c1);
        float4 b3 = *(const float4*)(qrow + 32 + c1 + 4);
        aQ1 = cat8(cvt8(b0, b1), cvt8(b2, b3));
    }

    const int ktiles = blockIdx.x * 4 + 4;   // causal frontier: k < q0+128

    // lane-local softmax state (exp2 domain) for 8 rows (row = r + 8*half)
    float m8[8], l8[8];
    #pragma unroll
    for (int r = 0; r < 8; ++r) { m8[r] = -3.0e38f; l8[r] = 0.0f; }

    const int kr = tid >> 3;          // staged k row 0..31
    const int d0 = (tid & 7) << 3;    // 8-wide d chunk

    // ================= PASS 1: softmax statistics (lane-local) =============
    for (int kt = 0; kt < ktiles; ++kt) {
        const int kbase = kt * 32;
        const int buf = kt & 1;
        cvt_store8(&sK[buf][kr][d0], Kp + (size_t)(kbase + kr) * D_DIM + d0);
        if (kt + 1 < ktiles)   // hide next tile's latency (global_prefetch_b8)
            __builtin_prefetch(Kp + (size_t)(kbase + 32 + kr) * D_DIM + d0, 0, 1);
        __syncthreads();

        #pragma unroll
        for (int kc = 0; kc < 32; kc += 16) {
            if (kbase + kc <= qlast) {           // scalar branch
                const _Float16* krow = &sK[buf][kc + n16][0];
                v16h b0 = cat8(*(const v8h*)(krow + 16 * half),
                               *(const v8h*)(krow + 16 * half + 8));
                v16h b1 = cat8(*(const v8h*)(krow + 32 + 16 * half),
                               *(const v8h*)(krow + 32 + 16 * half + 8));
                v8f acc = {};
                acc = __builtin_amdgcn_wmma_f32_16x16x32_f16(false, aQ0, false, b0,
                                                             (short)0, acc, false, false);
                acc = __builtin_amdgcn_wmma_f32_16x16x32_f16(false, aQ1, false, b1,
                                                             (short)0, acc, false, false);
                const int kg = kbase + kc + n16;
                #pragma unroll
                for (int r = 0; r < 8; ++r) {
                    const int qg = qrow0 + r + 8 * half;
                    const float s = (kg > qg) ? NEGBIG : acc[r] * SCALE2;
                    const float mn = fmaxf(m8[r], s);
                    l8[r] = l8[r] * __builtin_amdgcn_exp2f(m8[r] - mn)
                          + __builtin_amdgcn_exp2f(s - mn);
                    m8[r] = mn;
                }
            }
        }
    }
    __syncthreads();

    // merge the 16 lane-local softmax states per row (once)
    #pragma unroll
    for (int r = 0; r < 8; ++r) {
        #pragma unroll
        for (int off = 1; off < 16; off <<= 1) {
            const float om = __shfl_xor(m8[r], off, 32);
            const float ol = __shfl_xor(l8[r], off, 32);
            const float mn = fmaxf(m8[r], om);
            l8[r] = l8[r] * __builtin_amdgcn_exp2f(m8[r] - mn)
                  + ol    * __builtin_amdgcn_exp2f(om   - mn);
            m8[r] = mn;
        }
    }
    float linv[8];
    #pragma unroll
    for (int r = 0; r < 8; ++r) linv[r] = __builtin_amdgcn_rcpf(l8[r]);

    // ================= PASS 2: write P once, accumulate context =============
    v8f c0 = {}, c1 = {}, c2 = {}, c3 = {};
    for (int kt = 0; kt < ktiles; ++kt) {
        const int kbase = kt * 32;
        const int buf = kt & 1;
        cvt_store8(&sK[buf][kr][d0], Kp + (size_t)(kbase + kr) * D_DIM + d0);
        {   // stage V transposed: sVt[d][k]
            const float* src = Vp + (size_t)(kbase + kr) * D_DIM + d0;
            float4 a = ((const float4*)src)[0];
            float4 b = ((const float4*)src)[1];
            sVt[buf][d0 + 0][kr] = (_Float16)a.x; sVt[buf][d0 + 1][kr] = (_Float16)a.y;
            sVt[buf][d0 + 2][kr] = (_Float16)a.z; sVt[buf][d0 + 3][kr] = (_Float16)a.w;
            sVt[buf][d0 + 4][kr] = (_Float16)b.x; sVt[buf][d0 + 5][kr] = (_Float16)b.y;
            sVt[buf][d0 + 6][kr] = (_Float16)b.z; sVt[buf][d0 + 7][kr] = (_Float16)b.w;
        }
        if (kt + 1 < ktiles) {
            __builtin_prefetch(Kp + (size_t)(kbase + 32 + kr) * D_DIM + d0, 0, 1);
            __builtin_prefetch(Vp + (size_t)(kbase + 32 + kr) * D_DIM + d0, 0, 1);
        }
        __syncthreads();

        #pragma unroll
        for (int kc = 0; kc < 32; kc += 16) {
            if (kbase + kc <= qlast) {
                const _Float16* krow = &sK[buf][kc + n16][0];
                v16h b0 = cat8(*(const v8h*)(krow + 16 * half),
                               *(const v8h*)(krow + 16 * half + 8));
                v16h b1 = cat8(*(const v8h*)(krow + 32 + 16 * half),
                               *(const v8h*)(krow + 32 + 16 * half + 8));
                v8f acc = {};
                acc = __builtin_amdgcn_wmma_f32_16x16x32_f16(false, aQ0, false, b0,
                                                             (short)0, acc, false, false);
                acc = __builtin_amdgcn_wmma_f32_16x16x32_f16(false, aQ1, false, b1,
                                                             (short)0, acc, false, false);
                const int kg = kbase + kc + n16;
                #pragma unroll
                for (int r = 0; r < 8; ++r) {
                    const int qg  = qrow0 + r + 8 * half;
                    const int row = r + 8 * half;
                    const float s = (kg > qg) ? NEGBIG : acc[r] * SCALE2;
                    float p = __builtin_amdgcn_exp2f(s - m8[r]) * linv[r];
                    p = (kg > qg) ? 0.0f : p;
                    probp[(size_t)(qrow0 + row) * S_LEN + kg] = p;
                    sP[wave][row][kc + n16] = (_Float16)p;
                }
            } else {
                // fully masked: contiguous zero spans (lane -> row n16)
                float* dst = probp + (size_t)(qrow0 + n16) * S_LEN
                           + kbase + kc + 8 * half;
                const float4 z = make_float4(0.f, 0.f, 0.f, 0.f);
                ((float4*)dst)[0] = z;
                ((float4*)dst)[1] = z;
                v8h zh = {};
                *(v8h*)&sP[wave][n16][kc + 8 * half] = zh;
            }
        }

        if (kbase <= qlast) {       // scalar branch: any prob mass this tile?
            const _Float16* prow = &sP[wave][n16][0];
            v16h aP = cat8(*(const v8h*)(prow + 16 * half),
                           *(const v8h*)(prow + 16 * half + 8));
            #pragma unroll
            for (int t = 0; t < 4; ++t) {
                const _Float16* vrow = &sVt[buf][t * 16 + n16][0];
                v16h bv = cat8(*(const v8h*)(vrow + 16 * half),
                               *(const v8h*)(vrow + 16 * half + 8));
                v8f* cc = (t == 0) ? &c0 : (t == 1) ? &c1 : (t == 2) ? &c2 : &c3;
                *cc = __builtin_amdgcn_wmma_f32_16x16x32_f16(false, aP, false, bv,
                                                             (short)0, *cc, false, false);
            }
        }
    }

    // ---- context store: lane holds (row = r+8*half, d = t*16 + n16) ----
    #pragma unroll
    for (int t = 0; t < 4; ++t) {
        const v8f cc = (t == 0) ? c0 : (t == 1) ? c1 : (t == 2) ? c2 : c3;
        #pragma unroll
        for (int r = 0; r < 8; ++r) {
            const int row = r + 8 * half;
            ctxp[(size_t)(qrow0 + row) * D_DIM + t * 16 + n16] = cc[r];
        }
    }

    // ---- zero-fill attn_prob beyond the block's causal frontier ----
    const int zstart = q0 + 128;
    const int zlen4  = (S_LEN - zstart) >> 2;
    if (zlen4 > 0) {
        const float4 z = make_float4(0.f, 0.f, 0.f, 0.f);
        for (int idx = tid; idx < 128 * zlen4; idx += 256) {
            const int row = idx / zlen4;
            const int c4  = idx - row * zlen4;
            float4* dst = (float4*)(probp + (size_t)(q0 + row) * S_LEN + zstart) + c4;
            *dst = z;
        }
    }
}

extern "C" void kernel_launch(void* const* d_in, const int* in_sizes, int n_in,
                              void* d_out, int out_size, void* d_ws, size_t ws_size,
                              hipStream_t stream) {
    (void)in_sizes; (void)n_in; (void)out_size; (void)d_ws; (void)ws_size;
    const float* Q = (const float*)d_in[0];
    const float* K = (const float*)d_in[1];
    const float* V = (const float*)d_in[2];
    // d_in[3] = attn_mask (causal) — reproduced analytically in-kernel.

    float* ctx  = (float*)d_out;
    float* prob = ctx + (size_t)B_DIM * H_DIM * S_LEN * D_DIM;

    dim3 grid(S_LEN / 128, B_DIM * H_DIM);
    sdpa_causal_wmma_kernel<<<grid, 256, 0, stream>>>(Q, K, V, ctx, prob);
}
